// MultiHeadAttention_88038239633816
// MI455X (gfx1250) — matrix operand order repaired and checked by reading, not compiled, not run on hardware
//
#include <hip/hip_runtime.h>
#include <hip/hip_bf16.h>

// ---------------------------------------------------------------- constants
#define SEQQ   2048
#define BATCH  2
#define HID    1024
#define HEADS  16
#define HDIM   64
#define MROWS  (SEQQ * BATCH)     // 4096 rows for all [S*B, HID] GEMMs

typedef __attribute__((ext_vector_type(16))) _Float16 v16h;
typedef __attribute__((ext_vector_type(8)))  _Float16 v8h;
typedef __attribute__((ext_vector_type(8)))  float    v8f;
typedef __attribute__((ext_vector_type(4)))  float    v4f;

static __device__ __forceinline__ v8f wmma16(v16h a, v16h b, v8f c) {
  // D = A(16x32 f16) x B(32x16 f16) + C(16x16 f32)
  return __builtin_amdgcn_wmma_f32_16x16x32_f16(
      /*neg_a=*/false, a, /*neg_b=*/false, b,
      /*c_mod=*/(short)0, c, /*reuse_a=*/false, /*reuse_b=*/false);
}

// A fragment (16x32 f16): lane m = L%16, half = L/16.
// elements 0..15 = K = half*16 + 0..15
static __device__ __forceinline__ v16h a_frag_f32(const float* rowp, int k0, int half) {
  const v4f* p0 = (const v4f*)(rowp + k0 + half * 16);
  const v4f* p1 = (const v4f*)(rowp + k0 + 8 + half * 16);
  v4f f0 = p0[0], f1 = p0[1], f2 = p1[0], f3 = p1[1];
  v16h a;
#pragma unroll
  for (int e = 0; e < 4; ++e) {
    a[e]      = (_Float16)f0[e];
    a[e + 4]  = (_Float16)f1[e];
    a[e + 8]  = (_Float16)f2[e];
    a[e + 12] = (_Float16)f3[e];
  }
  return a;
}

static __device__ __forceinline__ v16h a_frag_f16(const _Float16* rowp, int k0, int half) {
  v8h lo = *(const v8h*)(rowp + k0 + half * 16);
  v8h hi = *(const v8h*)(rowp + k0 + 8 + half * 16);
  v16h a;
#pragma unroll
  for (int e = 0; e < 8; ++e) { a[e] = lo[e]; a[e + 8] = hi[e]; }
  return a;
}

// B fragment (32x16 f16): lane n = L%16, element e <-> K = (L/16)*16 + e.
// Source row (fixed n) must be contiguous over K: 16 consecutive fp32.
static __device__ __forceinline__ v16h b_frag_f32(const float* rowp, int k0, int half) {
  const v4f* p = (const v4f*)(rowp + k0 + half * 16);
  v4f w0 = p[0], w1 = p[1], w2 = p[2], w3 = p[3];
  v16h b;
#pragma unroll
  for (int e = 0; e < 4; ++e) {
    b[e]      = (_Float16)w0[e];
    b[e + 4]  = (_Float16)w1[e];
    b[e + 8]  = (_Float16)w2[e];
    b[e + 12] = (_Float16)w3[e];
  }
  return b;
}

// ---------------------------------------------------------------- linear: Y = X @ W^T + bias
// X: [MROWS, HID] (fp32 or f16), W: [HID, HID] fp32 row-major (W^T column n ==
// W row n, contiguous in K).  Each wave -> 16(M) x 64(N) output tile.
// Outputs (any may be null): Yh f16 [M,HID]; Yt f16 transposed [B,H,D,S]; Yf fp32 [M,HID].
template <int AH>
__global__ __launch_bounds__(256) void linear_kernel(
    const float* __restrict__ Xf, const _Float16* __restrict__ Xh,
    const float* __restrict__ W, const float* __restrict__ bias,
    _Float16* __restrict__ Yh, _Float16* __restrict__ Yt, float* __restrict__ Yf) {
  const int lane = threadIdx.x & 31;
  const int wave = threadIdx.x >> 5;
  const int quad = blockIdx.x * 8 + wave;          // 4096 total
  const int NT4  = HID / 64;                       // 16
  const int tm   = quad / NT4;
  const int tn   = quad % NT4;
  const int row0 = tm * 16;
  const int col0 = tn * 64;
  const int m    = lane & 15;
  const int half = lane >> 4;

  v8f acc[4];
#pragma unroll
  for (int t = 0; t < 4; ++t) acc[t] = (v8f){0.f, 0.f, 0.f, 0.f, 0.f, 0.f, 0.f, 0.f};

  const int arow = row0 + m;
  for (int k0 = 0; k0 < HID; k0 += 32) {
    v16h a;
    if constexpr (AH) a = a_frag_f16(Xh + (size_t)arow * HID, k0, half);
    else              a = a_frag_f32(Xf + (size_t)arow * HID, k0, half);
#pragma unroll
    for (int t = 0; t < 4; ++t) {
      const int nrow = col0 + t * 16 + m;
      v16h b = b_frag_f32(W + (size_t)nrow * HID, k0, half);
      acc[t] = wmma16(a, b, acc[t]);
    }
  }

#pragma unroll
  for (int t = 0; t < 4; ++t) {
    const int col = col0 + t * 16 + m;             // C/D: N = lane%16
    const float bv = bias[col];
#pragma unroll
    for (int r = 0; r < 8; ++r) {
      const int row = row0 + r + half * 8;         // C/D: M = r + 8*(lane/16)
      const float val = acc[t][r] + bv;
      if (Yh) Yh[(size_t)row * HID + col] = (_Float16)val;
      if (Yf) Yf[(size_t)row * HID + col] = val;
      if (Yt) {                                    // [B,H,D,S] f16 (for PV B-operand)
        const int bb = row % BATCH;
        const int s  = row / BATCH;
        const int h  = col / HDIM;
        const int d  = col % HDIM;
        Yt[(((size_t)bb * HEADS + h) * HDIM + d) * SEQQ + s] = (_Float16)val;
      }
    }
  }
}

// ---------------------------------------------------------------- fused scores + softmax + PV
// Block = (b, h, 16 query rows).  Wave w owns key columns [w*256, w*256+256):
// 16 prob tiles live in registers; block-wide softmax via shfl + LDS; probs are
// written to the mandatory attn output exactly once and NEVER re-read from HBM:
// PV (ctx = P @ V) is fused here via an LDS transpose (C/D layout -> A layout)
// processed in 2 phases of 128 columns, with cross-wave reduction via ds_add_f32.
__global__ __launch_bounds__(256) void attn_fused_kernel(
    const _Float16* __restrict__ qh, const _Float16* __restrict__ kh,
    const _Float16* __restrict__ vt, float* __restrict__ attn,
    _Float16* __restrict__ ctx) {
  const int lane = threadIdx.x & 31;
  const int wave = threadIdx.x >> 5;
  const int m    = lane & 15;
  const int half = lane >> 4;

  int blk = blockIdx.x;                            // b*H*128 + h*128 + tq
  const int tq = blk % (SEQQ / 16); blk /= (SEQQ / 16);
  const int h  = blk % HEADS;
  const int b  = blk / HEADS;

  __shared__ float    red[8][16];
  __shared__ float    rowv[16];
  __shared__ _Float16 plds[8][16][136];            // per-wave 16x128 prob tile, padded rows
  __shared__ float    cpart[16][64];               // ctx reduction tile

  // zero the ctx reduction tile (visible after the first softmax barrier)
  for (int i = threadIdx.x; i < 16 * 64; i += 256) ((float*)cpart)[i] = 0.f;

  // ---------------- scores: S = Q K^T / sqrt(D)
  const int sq = tq * 16 + m;                      // A row for this lane
  const _Float16* qrow  = qh + ((size_t)sq * BATCH + b) * HID + h * HDIM;
  const _Float16* vbase = vt + ((size_t)b * HEADS + h) * HDIM * SEQQ;

  v8f acc[16];
#pragma unroll
  for (int t = 0; t < 16; ++t) acc[t] = (v8f){0.f, 0.f, 0.f, 0.f, 0.f, 0.f, 0.f, 0.f};

#pragma unroll
  for (int k0 = 0; k0 < HDIM; k0 += 32) {
    v16h a = a_frag_f16(qrow, k0, half);
#pragma unroll
    for (int t = 0; t < 16; ++t) {
      const int sk = wave * 256 + t * 16 + m;      // B: N = lane%16
      v16h bf = *(const v16h*)(kh + ((size_t)sk * BATCH + b) * HID + h * HDIM + k0 + half * 16);
      acc[t] = wmma16(a, bf, acc[t]);
    }
  }

  const float scale = 0.125f;                      // 1/sqrt(64)
#pragma unroll
  for (int t = 0; t < 16; ++t)
#pragma unroll
    for (int r = 0; r < 8; ++r) acc[t][r] *= scale;

  // ---------------- softmax: row max
  float pm[8];
#pragma unroll
  for (int r = 0; r < 8; ++r) {
    float v = -3.0e38f;
#pragma unroll
    for (int t = 0; t < 16; ++t) v = fmaxf(v, acc[t][r]);
#pragma unroll
    for (int mask = 8; mask >= 1; mask >>= 1) v = fmaxf(v, __shfl_xor(v, mask));
    pm[r] = v;
  }
  if (m == 0) {
#pragma unroll
    for (int r = 0; r < 8; ++r) red[wave][half * 8 + r] = pm[r];
  }
  __syncthreads();
  if (threadIdx.x < 16) {
    float v = -3.0e38f;
#pragma unroll
    for (int w = 0; w < 8; ++w) v = fmaxf(v, red[w][threadIdx.x]);
    rowv[threadIdx.x] = v;
  }
  __syncthreads();
  float rmax[8];
#pragma unroll
  for (int r = 0; r < 8; ++r) rmax[r] = rowv[half * 8 + r];

  // ---------------- softmax: exp + row sum
  float ps[8];
#pragma unroll
  for (int r = 0; r < 8; ++r) ps[r] = 0.f;
#pragma unroll
  for (int t = 0; t < 16; ++t)
#pragma unroll
    for (int r = 0; r < 8; ++r) {
      float e = __expf(acc[t][r] - rmax[r]);
      acc[t][r] = e;
      ps[r] += e;
    }
#pragma unroll
  for (int r = 0; r < 8; ++r) {
    float v = ps[r];
#pragma unroll
    for (int mask = 8; mask >= 1; mask >>= 1) v += __shfl_xor(v, mask);
    ps[r] = v;
  }
  if (m == 0) {
#pragma unroll
    for (int r = 0; r < 8; ++r) red[wave][half * 8 + r] = ps[r];
  }
  __syncthreads();
  if (threadIdx.x < 16) {
    float v = 0.f;
#pragma unroll
    for (int w = 0; w < 8; ++w) v += red[w][threadIdx.x];
    rowv[threadIdx.x] = v;
  }
  __syncthreads();
  float rinv[8];
#pragma unroll
  for (int r = 0; r < 8; ++r) rinv[r] = 1.0f / rowv[half * 8 + r];

  // ---------------- fused prob write + PV (2 phases x 128 columns per wave)
  float* abase = attn + ((size_t)b * HEADS + h) * SEQQ * SEQQ;
  v8f cacc[4];
#pragma unroll
  for (int t = 0; t < 4; ++t) cacc[t] = (v8f){0.f, 0.f, 0.f, 0.f, 0.f, 0.f, 0.f, 0.f};

#pragma unroll
  for (int p = 0; p < 2; ++p) {
    // write this phase's probs: once to HBM (mandatory output), once to LDS (f16)
#pragma unroll
    for (int t = p * 8; t < p * 8 + 8; ++t) {
      const int sk = wave * 256 + t * 16 + m;      // global key column
      const int lc = t * 16 + m - p * 128;         // local column 0..127
#pragma unroll
      for (int r = 0; r < 8; ++r) {
        const int rr = r + half * 8;               // local query row 0..15
        const float pr = acc[t][r] * rinv[r];
        abase[(size_t)(tq * 16 + rr) * SEQQ + sk] = pr;
        plds[wave][rr][lc] = (_Float16)pr;
      }
    }
    __syncthreads();
    // PV: A = probs (LDS, row-major local), B = vt (contiguous over sk)
#pragma unroll
    for (int kk = 0; kk < 128; kk += 32) {
      v16h a = a_frag_f16(&plds[wave][m][0], kk, half);
      const int kg = wave * 256 + p * 128 + kk;    // global sk base
#pragma unroll
      for (int t = 0; t < 4; ++t) {
        const int d = t * 16 + m;
        v16h bf = *(const v16h*)(vbase + (size_t)d * SEQQ + kg + half * 16);
        cacc[t] = wmma16(a, bf, cacc[t]);
      }
    }
    __syncthreads();                               // before next phase overwrites plds
  }

  // ---------------- cross-wave ctx reduction (ds_add_f32) and store
#pragma unroll
  for (int t = 0; t < 4; ++t)
#pragma unroll
    for (int r = 0; r < 8; ++r)
      atomicAdd(&cpart[r + half * 8][t * 16 + m], cacc[t][r]);
  __syncthreads();
  for (int i = threadIdx.x; i < 16 * 64; i += 256) {
    const int rr = i >> 6, dd = i & 63;
    const int sqr = tq * 16 + rr;
    ctx[((size_t)sqr * BATCH + b) * HID + h * HDIM + dd] = (_Float16)cpart[rr][dd];
  }
}

// ---------------------------------------------------------------- launcher
extern "C" void kernel_launch(void* const* d_in, const int* in_sizes, int n_in,
                              void* d_out, int out_size, void* d_ws, size_t ws_size,
                              hipStream_t stream) {
  (void)in_sizes; (void)n_in; (void)out_size; (void)ws_size;
  const float* query = (const float*)d_in[0];
  const float* key_  = (const float*)d_in[1];
  const float* value = (const float*)d_in[2];
  const float* Wq = (const float*)d_in[3];
  const float* bq = (const float*)d_in[4];
  const float* Wk = (const float*)d_in[5];
  const float* bk = (const float*)d_in[6];
  const float* Wv = (const float*)d_in[7];
  const float* bv = (const float*)d_in[8];
  const float* Wo = (const float*)d_in[9];
  const float* bo = (const float*)d_in[10];

  float* out  = (float*)d_out;                         // [S,B,HID]
  float* attn = out + (size_t)MROWS * HID;             // [B,H,S,S]

  _Float16* qh  = (_Float16*)d_ws;                     // 8 MB each
  _Float16* kh  = qh + (size_t)MROWS * HID;
  _Float16* vt  = kh + (size_t)MROWS * HID;            // transposed [B,H,D,S]
  _Float16* ctx = vt + (size_t)MROWS * HID;

  dim3 blk(256);
  dim3 glin(512);                                      // 4096 waves, 16x64 tiles

  linear_kernel<0><<<glin, blk, 0, stream>>>(query, nullptr, Wq, bq, qh, nullptr, nullptr);
  linear_kernel<0><<<glin, blk, 0, stream>>>(key_,  nullptr, Wk, bk, kh, nullptr, nullptr);
  linear_kernel<0><<<glin, blk, 0, stream>>>(value, nullptr, Wv, bv, nullptr, vt, nullptr);

  attn_fused_kernel<<<dim3(BATCH * HEADS * (SEQQ / 16)), blk, 0, stream>>>(qh, kh, vt, attn, ctx);

  linear_kernel<1><<<glin, blk, 0, stream>>>(nullptr, ctx, Wo, bo, nullptr, nullptr, out);
}
